// ManifoldHyperConnect_23862838297292
// MI455X (gfx1250) — matrix operand order repaired and checked by reading, not compiled, not run on hardware
//
#include <hip/hip_runtime.h>
#include <hip/hip_bf16.h>

typedef __attribute__((ext_vector_type(16))) _Float16 v16h;
typedef __attribute__((ext_vector_type(8)))  float    v8f;

#define NSTREAM 4
#define DIM 1024
#define NC 4096            // NSTREAM * DIM
#define TOK_PER_BLK 16
#define THREADS 256
#define WAVES 8
#define SINKHORN_ITERS 20
#define RMS_EPS 1.1920929e-7f

__device__ __forceinline__ float sigmoid_f(float v) {
    return 1.0f / (1.0f + __expf(-v));
}

__global__ __launch_bounds__(THREADS)
void mhc_fused_kernel(const float* __restrict__ x,
                      const float* __restrict__ rms_w,
                      const float* __restrict__ phi_pre,
                      const float* __restrict__ phi_post,
                      const float* __restrict__ phi_res,
                      const float* __restrict__ b_pre,
                      const float* __restrict__ b_post,
                      const float* __restrict__ b_res,
                      const float* __restrict__ alpha_pre_p,
                      const float* __restrict__ alpha_post_p,
                      const float* __restrict__ alpha_res_p,
                      float* __restrict__ out)
{
    __shared__ float s_red[WAVES][16][32];     // per-wave partial C tiles (split-K)
    __shared__ float s_ssq[WAVES][TOK_PER_BLK];// per-wave partial sum-of-squares
    __shared__ float s_raw[TOK_PER_BLK][24];
    __shared__ float s_G[TOK_PER_BLK][16];     // fused 4x4 mixing matrix per token

    const int tid  = threadIdx.x;
    const int tok0 = blockIdx.x * TOK_PER_BLK;

    // --- Phase 1: WMMA gate projections raw[16 tok x 24], split-K over 8 waves.
    //     RMS sum-of-squares is folded into the A-fragment loads (x read once).
    {
        const int wave = tid >> 5;
        const int lane = tid & 31;
        const int c    = lane & 15;      // A: M row (token); B/C: N column
        const int hi   = lane >> 4;

        const float* xrow = x + (size_t)(tok0 + c) * NC;
        // combined weight matrix W[24][NC]: rows 0-3 phi_pre, 4-7 phi_post, 8-23 phi_res
        const float* w0row = (c < 4) ? (phi_pre + c * NC)
                           : (c < 8) ? (phi_post + (c - 4) * NC)
                                     : (phi_res + (c - 8) * NC);
        // tile-1 columns c>=8 are never consumed downstream -> clamp to a valid
        // row instead of branching/masking (D columns are independent).
        const float* w1row = phi_res + (size_t)(((c < 8) ? (8 + c) : 8)) * NC;

        v8f acc0 = {};
        v8f acc1 = {};
        float ss = 0.0f;
        const int kbase = wave * (NC / WAVES);     // 512-wide K slab per wave
        #pragma unroll 2
        for (int kc = 0; kc < (NC / WAVES) / 32; ++kc) {
            const int K0 = kbase + kc * 32;
            __builtin_prefetch(xrow + K0 + 64, 0, 3);   // global_prefetch_b8
            v16h a, b0, b1;
            #pragma unroll
            for (int p = 0; p < 8; ++p) {
                // A f16 16x32 layout: lane<16 -> K {0..7,16..23}; lane>=16 -> {8..15,24..31}
                const int ka = ((p >> 2) << 4) + (hi << 3) + ((p & 3) << 1);
                float2 xv = *(const float2*)(xrow + K0 + ka);
                float2 wv = *(const float2*)(rms_w + K0 + ka);
                ss += xv.x * xv.x + xv.y * xv.y;            // RMS stats for free
                a[2 * p]     = (_Float16)(xv.x * wv.x);
                a[2 * p + 1] = (_Float16)(xv.y * wv.y);
                // B f16 32x16 layout: lane<16 -> K {0..7,16..23}; lane>=16 -> {8..15,24..31} (N = lane%16)
                const int kb = ((p >> 2) << 4) + (hi << 3) + ((p & 3) << 1);
                float2 w0 = *(const float2*)(w0row + K0 + kb);
                float2 w1 = *(const float2*)(w1row + K0 + kb);
                b0[2 * p]     = (_Float16)w0.x;
                b0[2 * p + 1] = (_Float16)w0.y;
                b1[2 * p]     = (_Float16)w1.x;
                b1[2 * p + 1] = (_Float16)w1.y;
            }
            acc0 = __builtin_amdgcn_wmma_f32_16x16x32_f16(
                       false, a, false, b0, (short)0, acc0, false, false);
            acc1 = __builtin_amdgcn_wmma_f32_16x16x32_f16(
                       false, a, false, b1, (short)0, acc1, false, false);
        }
        // combine the lane pair (c, c+16): together they cover this wave's K slab
        ss += __shfl_xor(ss, 16, 32);
        if (hi == 0) s_ssq[wave][c] = ss;
        // C layout: VGPR r -> M = r + 8*hi, N = lane%16
        #pragma unroll
        for (int r = 0; r < 8; ++r) {
            s_red[wave][r + hi * 8][c]      = acc0[r];
            s_red[wave][r + hi * 8][16 + c] = acc1[r];
        }
    }
    __syncthreads();

    // ------ Phase 2: cross-wave K reduction + RMS scale + alpha/bias ------------
    {
        const float a_pre  = alpha_pre_p[0];
        const float a_post = alpha_post_p[0];
        const float a_res  = alpha_res_p[0];
        for (int e = tid; e < TOK_PER_BLK * 24; e += THREADS) {
            const int row = e / 24;
            const int col = e % 24;
            float s  = 0.0f;
            float sq = 0.0f;
            #pragma unroll
            for (int w = 0; w < WAVES; ++w) {
                s  += s_red[w][row][col];
                sq += s_ssq[w][row];
            }
            const float ms    = sq * (1.0f / (float)NC);
            const float scale = rsqrtf(ms + RMS_EPS);
            float alpha, bias;
            if (col < 4)      { alpha = a_pre;  bias = b_pre[col]; }
            else if (col < 8) { alpha = a_post; bias = b_post[col - 4]; }
            else              { alpha = a_res;  bias = b_res[col - 8]; }
            s_raw[row][col] = alpha * scale * s + bias;
        }
    }
    __syncthreads();

    // ------ Phase 3: sigmoid gates + Sinkhorn-Knopp, fold into 4x4 G ------------
    if (tid < TOK_PER_BLK) {
        const int t = tid;
        float Hpre[4], Hpost[4], M[16];
        #pragma unroll
        for (int n = 0; n < 4; ++n) {
            Hpre[n]  = sigmoid_f(s_raw[t][n]);
            Hpost[n] = 2.0f * sigmoid_f(s_raw[t][4 + n]);
        }
        #pragma unroll
        for (int k = 0; k < 16; ++k) M[k] = __expf(s_raw[t][8 + k]);
        for (int it = 0; it < SINKHORN_ITERS; ++it) {
            #pragma unroll
            for (int i = 0; i < 4; ++i) {
                const float inv = 1.0f /
                    (M[i * 4] + M[i * 4 + 1] + M[i * 4 + 2] + M[i * 4 + 3] + 1e-8f);
                M[i * 4] *= inv; M[i * 4 + 1] *= inv;
                M[i * 4 + 2] *= inv; M[i * 4 + 3] *= inv;
            }
            #pragma unroll
            for (int j = 0; j < 4; ++j) {
                const float inv = 1.0f /
                    (M[j] + M[4 + j] + M[8 + j] + M[12 + j] + 1e-8f);
                M[j] *= inv; M[4 + j] *= inv; M[8 + j] *= inv; M[12 + j] *= inv;
            }
        }
        // out[i] = sum_j (Hres[i][j] + Hpost[i]*Hpre[j]) * x[j]
        #pragma unroll
        for (int i = 0; i < 4; ++i)
            #pragma unroll
            for (int j = 0; j < 4; ++j)
                s_G[t][i * 4 + j] = M[i * 4 + j] + Hpost[i] * Hpre[j];
    }
    __syncthreads();

    // ------ Phase 4: apply fused 4x4 mix, streamed float4 in/out ----------------
    {
        const int t = tid >> 4;
        const int j = tid & 15;
        const float* xt = x   + (size_t)(tok0 + t) * NC;
        float*       ot = out + (size_t)(tok0 + t) * NC;
        float G[16];
        #pragma unroll
        for (int k = 0; k < 16; ++k) G[k] = s_G[t][k];
        for (int db = 0; db < DIM; db += 64) {
            const int d = db + j * 4;
            const float4 x0 = *(const float4*)(xt + 0 * DIM + d);
            const float4 x1 = *(const float4*)(xt + 1 * DIM + d);
            const float4 x2 = *(const float4*)(xt + 2 * DIM + d);
            const float4 x3 = *(const float4*)(xt + 3 * DIM + d);
            #pragma unroll
            for (int i = 0; i < 4; ++i) {
                float4 o;
                o.x = G[i*4+0]*x0.x + G[i*4+1]*x1.x + G[i*4+2]*x2.x + G[i*4+3]*x3.x;
                o.y = G[i*4+0]*x0.y + G[i*4+1]*x1.y + G[i*4+2]*x2.y + G[i*4+3]*x3.y;
                o.z = G[i*4+0]*x0.z + G[i*4+1]*x1.z + G[i*4+2]*x2.z + G[i*4+3]*x3.z;
                o.w = G[i*4+0]*x0.w + G[i*4+1]*x1.w + G[i*4+2]*x2.w + G[i*4+3]*x3.w;
                *(float4*)(ot + i * DIM + d) = o;
            }
        }
    }
}

extern "C" void kernel_launch(void* const* d_in, const int* in_sizes, int n_in,
                              void* d_out, int out_size, void* d_ws, size_t ws_size,
                              hipStream_t stream) {
    const float* x          = (const float*)d_in[0];
    const float* rms_w      = (const float*)d_in[1];
    const float* phi_pre    = (const float*)d_in[2];
    const float* phi_post   = (const float*)d_in[3];
    const float* phi_res    = (const float*)d_in[4];
    const float* b_pre      = (const float*)d_in[5];
    const float* b_post     = (const float*)d_in[6];
    const float* b_res      = (const float*)d_in[7];
    const float* alpha_pre  = (const float*)d_in[8];
    const float* alpha_post = (const float*)d_in[9];
    const float* alpha_res  = (const float*)d_in[10];
    float* out = (float*)d_out;

    const int total_tokens = in_sizes[0] / NC;           // B*S = 16384
    const int blocks = total_tokens / TOK_PER_BLK;       // 1024

    mhc_fused_kernel<<<blocks, THREADS, 0, stream>>>(
        x, rms_w, phi_pre, phi_post, phi_res,
        b_pre, b_post, b_res,
        alpha_pre, alpha_post, alpha_res, out);
}
